// JointModel_46823733461791
// MI455X (gfx1250) — matrix operand order repaired and checked by reading, not compiled, not run on hardware
//
#include <hip/hip_runtime.h>
#include <hip/hip_bf16.h>
#include <math.h>

// ---------------------------------------------------------------------------
// Problem constants (from the reference)
// ---------------------------------------------------------------------------
#define NP    2048
#define NW    89
#define NALL  2137
#define B_SZ  16384
#define E2D   768
#define UWD   313
#define NHEAD 8
#define HDIM  32

typedef __attribute__((ext_vector_type(16))) __bf16 v16bf;
typedef __attribute__((ext_vector_type(8)))  float  v8f;

// ---------------------------------------------------------------------------
// Generic batched GEMM: C[z] = act(A[z] (MxK) @ B[z] (KxN) + bias)
// bf16 WMMA (v_wmma_f32_16x16x32_bf16), f32 accumulate.
//  - one wave per 16(M) x 32(N) C strip: two WMMAs sharing one A fragment
//  - GEMM_WAVES=4 waves per block along M; B tile (32K x 32N) staged in LDS in
//    WMMA-fragment order: one coalesced float4 per thread per fragment column
//    half (128 threads * 8 elements = 1024), then one aligned 32B ds_load per
//    lane per fragment
//  - A fragment: two unguarded contiguous 8-float runs per lane (k = 16*khalf + e, same order as B),
//    compiler emits a clause of global_load_b128
//  - main loop over K&~31 is guard-free; single guarded tail handles K%32
//  - row clamp instead of early exit so __syncthreads stays uniform
// Requirements: N % 32 == 0, ldB % 4 == 0, B base 16B aligned (all call sites)
// act: 0=none, 1=relu, 2=elu
// ---------------------------------------------------------------------------
#define GEMM_WAVES 4

__global__ void gemm_bf16_wmma(const float* __restrict__ A,
                               const float* __restrict__ Bm,
                               const float* __restrict__ bias,
                               float* __restrict__ C,
                               int M, int K, int N,
                               int ldA, int ldB, int ldC,
                               long sA, long sB, long sC, int act) {
  __shared__ v16bf sBf[2][32];               // 32Kx32N bf16 B tile, fragment order

  const int tid   = threadIdx.x;
  const int lane  = tid & 31;
  const int wave  = tid >> 5;
  const int mtile = blockIdx.x * GEMM_WAVES + wave;
  const int m0    = mtile * 16;
  const int n0    = blockIdx.y * 32;

  const float* Ab = A  + (long)blockIdx.z * sA;
  const float* Bb = Bm + (long)blockIdx.z * sB;
  float*       Cb = C  + (long)blockIdx.z * sC;

  const int row   = lane & 15;               // A: row within tile / B,C: column
  const int khalf = lane >> 4;               // half-lane group selects K half
  int m = m0 + row;
  if (m > M - 1) m = M - 1;                  // clamp loads; stores are guarded
  const float* arow = Ab + (long)m * ldA;

  // B-staging addressing (static blockDim = 128): thread t loads float4 from
  // tile row kk = t>>2, columns nn4..nn4+3 of each 16-column fragment half.
  const int skk  = tid >> 2;                 // 0..31 (K within tile)
  const int snn4 = (tid & 3) << 2;           // 0,4,8,12 (N within fragment)
  const int kidx = ((skk & 16) << 4) + (skk & 15);  // fragment linear base
  __bf16* s0 = (__bf16*)&sBf[0][0];
  __bf16* s1 = (__bf16*)&sBf[1][0];

  v8f acc0 = {}, acc1 = {};
  const int K32 = K & ~31;

  for (int k0 = 0; k0 < K32; k0 += 32) {
    // ---- stage 32x32 B tile into LDS in fragment order -------------------
    __syncthreads();                         // protect previous iter's reads
    {
      const float* bp = Bb + (long)(k0 + skk) * ldB + n0;
      const float4 q0 = *(const float4*)(bp + snn4);
      const float4 q1 = *(const float4*)(bp + 16 + snn4);
      s0[kidx + ((snn4 + 0) << 4)] = (__bf16)q0.x;
      s0[kidx + ((snn4 + 1) << 4)] = (__bf16)q0.y;
      s0[kidx + ((snn4 + 2) << 4)] = (__bf16)q0.z;
      s0[kidx + ((snn4 + 3) << 4)] = (__bf16)q0.w;
      s1[kidx + ((snn4 + 0) << 4)] = (__bf16)q1.x;
      s1[kidx + ((snn4 + 1) << 4)] = (__bf16)q1.y;
      s1[kidx + ((snn4 + 2) << 4)] = (__bf16)q1.z;
      s1[kidx + ((snn4 + 3) << 4)] = (__bf16)q1.w;
    }
    __syncthreads();
    const v16bf b0 = sBf[0][lane];           // aligned 32B LDS fragment reads
    const v16bf b1 = sBf[1][lane];

    // ---- A fragment: two contiguous 8-float runs (k = 16*khalf + e, same order as B) ----
    v16bf af;
#pragma unroll
    for (int e = 0; e < 8; ++e)
      af[e] = (__bf16)arow[k0 + khalf * 16 + e];
#pragma unroll
    for (int e = 0; e < 8; ++e)
      af[e + 8] = (__bf16)arow[k0 + khalf * 16 + 8 + e];

    if (k0 + 32 < K32)                       // hint next A tile (global_prefetch)
      __builtin_prefetch(arow + k0 + 32, 0, 0);

    acc0 = __builtin_amdgcn_wmma_f32_16x16x32_bf16(
        false, af, false, b0, (short)0, acc0, false, false);
    acc1 = __builtin_amdgcn_wmma_f32_16x16x32_bf16(
        false, af, false, b1, (short)0, acc1, false, false);
  }

  // ---- guarded tail for K % 32 (runs at most once; K=2137 case) -----------
  if (K32 < K) {
    v16bf af, b0, b1;
#pragma unroll
    for (int e = 0; e < 16; ++e) {
      const int vg = e >> 1;
      const int ka = K32 + khalf * 16 + e;
      const int kb = K32 + khalf * 16 + e;
      af[e] = (__bf16)((ka < K) ? arow[ka] : 0.0f);
      b0[e] = (__bf16)((kb < K) ? Bb[(long)kb * ldB + (n0 + row)] : 0.0f);
      b1[e] = (__bf16)((kb < K) ? Bb[(long)kb * ldB + (n0 + 16 + row)] : 0.0f);
    }
    acc0 = __builtin_amdgcn_wmma_f32_16x16x32_bf16(
        false, af, false, b0, (short)0, acc0, false, false);
    acc1 = __builtin_amdgcn_wmma_f32_16x16x32_bf16(
        false, af, false, b1, (short)0, acc1, false, false);
  }

  // ---- epilogue: bias + activation, store with M guard ---------------------
#pragma unroll
  for (int i = 0; i < 8; ++i) {
    const int mm = m0 + khalf * 8 + i;       // C layout: VGPR i -> M=i (+8 hi lanes)
    if (mm < M) {
      const int nn0 = n0 + row;
      const int nn1 = n0 + 16 + row;
      float v0 = acc0[i], v1 = acc1[i];
      if (bias) { v0 += bias[nn0]; v1 += bias[nn1]; }
      if (act == 1) {
        v0 = fmaxf(v0, 0.0f); v1 = fmaxf(v1, 0.0f);
      } else if (act == 2) {                 // ELU(alpha=1)
        v0 = (v0 > 0.0f) ? v0 : (__expf(v0) - 1.0f);
        v1 = (v1 > 0.0f) ? v1 : (__expf(v1) - 1.0f);
      }
      Cb[(long)mm * ldC + nn0] = v0;
      Cb[(long)mm * ldC + nn1] = v1;
    }
  }
}

// ---------------------------------------------------------------------------
// e_src[h,n] = hp[h,n,:]·a_src[h,:] ; e_dst likewise
// ---------------------------------------------------------------------------
__global__ void edge_scores_kernel(const float* __restrict__ hp,
                                   const float* __restrict__ a_src,
                                   const float* __restrict__ a_dst,
                                   float* __restrict__ es,
                                   float* __restrict__ ed, int N) {
  const int i = blockIdx.x * blockDim.x + threadIdx.x;
  if (i >= NHEAD * N) return;
  const int h = i / N, n = i % N;
  const float* hpn = hp + ((long)h * N + n) * HDIM;
  const float* as  = a_src + h * HDIM;
  const float* ad  = a_dst + h * HDIM;
  float s = 0.0f, d = 0.0f;
#pragma unroll 8
  for (int o = 0; o < HDIM; ++o) { const float v = hpn[o]; s += v * as[o]; d += v * ad[o]; }
  es[i] = s;
  ed[i] = d;
}

// ---------------------------------------------------------------------------
// Masked leaky-relu softmax attention rows. One 256-thread block per row n;
// adjacency row staged once in LDS and reused across all 8 heads.
// attn[h,n,m] = softmax_m( adj[n,m]>0 ? lrelu(e_src[h,n]+e_dst[h,m]) : -1e9 )
// ---------------------------------------------------------------------------
__global__ void attn_softmax_kernel(const int* __restrict__ adj,
                                    const float* __restrict__ e_src,
                                    const float* __restrict__ e_dst,
                                    float* __restrict__ attn, int N) {
  const int rown = blockIdx.x;
  const int tid  = threadIdx.x;
  __shared__ int   s_adj[2144];
  __shared__ float s_red[256];

  for (int c = tid; c < N; c += 256) s_adj[c] = adj[(long)rown * N + c];
  __syncthreads();

  for (int h = 0; h < NHEAD; ++h) {
    const float esn = e_src[h * N + rown];
    float ev[9];                       // ceil(2137/256) = 9
    float lmax = -INFINITY;
    int cnt = 0;
    for (int c = tid; c < N; c += 256) {
      float e = esn + e_dst[h * N + c];
      e = (e >= 0.0f) ? e : 0.2f * e;               // leaky_relu(0.2)
      e = (s_adj[c] > 0) ? e : -1.0e9f;             // adjacency mask
      ev[cnt++] = e;
      lmax = fmaxf(lmax, e);
    }
    s_red[tid] = lmax; __syncthreads();
    for (int s = 128; s > 0; s >>= 1) {
      if (tid < s) s_red[tid] = fmaxf(s_red[tid], s_red[tid + s]);
      __syncthreads();
    }
    const float rmax = s_red[0]; __syncthreads();

    float lsum = 0.0f; cnt = 0;
    for (int c = tid; c < N; c += 256) {
      const float x = __expf(ev[cnt] - rmax);
      ev[cnt++] = x;
      lsum += x;
    }
    s_red[tid] = lsum; __syncthreads();
    for (int s = 128; s > 0; s >>= 1) {
      if (tid < s) s_red[tid] += s_red[tid + s];
      __syncthreads();
    }
    const float inv = 1.0f / s_red[0]; __syncthreads();

    cnt = 0;
    for (int c = tid; c < N; c += 256)
      attn[((long)h * N + rown) * N + c] = ev[cnt++] * inv;
  }
}

// ---------------------------------------------------------------------------
// Small helper kernels
// ---------------------------------------------------------------------------
__global__ void copy_words_kernel(const float* __restrict__ wf, float* __restrict__ h1wp) {
  const int i = blockIdx.x * blockDim.x + threadIdx.x;
  if (i < NW * 256) h1wp[i] = wf[i];
}

// mean over heads for the first NW rows: h2w[n,o] = mean_h h2heads[h,n,o]
__global__ void mean_words_kernel(const float* __restrict__ h2heads, float* __restrict__ h2w) {
  const int i = blockIdx.x * blockDim.x + threadIdx.x;
  if (i >= NW * HDIM) return;
  const int n = i / HDIM, o = i % HDIM;
  float s = 0.0f;
#pragma unroll
  for (int h = 0; h < NHEAD; ++h) s += h2heads[((long)h * NALL + n) * HDIM + o];
  h2w[i] = s * (1.0f / NHEAD);
}

// xraw[b,j] = sum_{w<89} uwa[b,w] * h2w[w,j]     (K=89 -> scalar)
__global__ void xraw_kernel(const float* __restrict__ uwa, const float* __restrict__ h2w,
                            float* __restrict__ xr) {
  const int i = blockIdx.x * blockDim.x + threadIdx.x;
  if (i >= B_SZ * HDIM) return;
  const int b = i / HDIM, j = i % HDIM;
  float s = 0.0f;
  for (int w = 0; w < NW; ++w) s += uwa[(long)b * UWD + w] * h2w[w * HDIM + j];
  xr[i] = s;
}

// x_c = [xraw(32) | uwa[:,89:](224)]; x = x_c / rowsum(x_c). Block per row.
__global__ void xnorm_kernel(const float* __restrict__ xr, const float* __restrict__ uwa,
                             float* __restrict__ xout) {
  const int b = blockIdx.x, t = threadIdx.x;
  __shared__ float sr[256];
  const float v = (t < HDIM) ? xr[(long)b * HDIM + t]
                             : uwa[(long)b * UWD + NW + (t - HDIM)];
  sr[t] = v; __syncthreads();
  for (int s = 128; s > 0; s >>= 1) {
    if (t < s) sr[t] += sr[t + s];
    __syncthreads();
  }
  xout[(long)b * 256 + t] = v / sr[0];
}

__global__ void concat_kernel(const float* __restrict__ a, const float* __restrict__ b,
                              float* __restrict__ out) {
  const int i = blockIdx.x * blockDim.x + threadIdx.x;
  if (i >= B_SZ * 256) return;
  const int r = i >> 8, c = i & 255;
  out[i] = (c < 128) ? a[(long)r * 128 + c] : b[(long)r * 128 + (c - 128)];
}

// x_temp = t(16384x128) @ W_fb(128x5) + b_fb  (N=5 -> scalar)
__global__ void final5_kernel(const float* __restrict__ t, const float* __restrict__ Wfb,
                              const float* __restrict__ bfb, float* __restrict__ out) {
  const int i = blockIdx.x * blockDim.x + threadIdx.x;
  if (i >= B_SZ * 5) return;
  const int b = i / 5, c = i % 5;
  float s = bfb[c];
  for (int k = 0; k < 128; ++k) s += t[(long)b * 128 + k] * Wfb[k * 5 + c];
  out[i] = s;
}

// ---------------------------------------------------------------------------
// Host-side orchestration
// ---------------------------------------------------------------------------
static inline dim3 g1(long n, int blk) { return dim3((unsigned)((n + blk - 1) / blk)); }

extern "C" void kernel_launch(void* const* d_in, const int* in_sizes, int n_in,
                              void* d_out, int out_size, void* d_ws, size_t ws_size,
                              hipStream_t stream) {
  (void)in_sizes; (void)n_in; (void)out_size; (void)ws_size;

  const float* pern_feature  = (const float*)d_in[0];
  const float* word_feature  = (const float*)d_in[1];
  const int*   pern_adj      = (const int*)  d_in[2];
  const int*   word_pern_adj = (const int*)  d_in[3];
  const float* user_word_adj = (const float*)d_in[4];
  const float* sentence_emb  = (const float*)d_in[5];
  const float* w1    = (const float*)d_in[6];
  const float* a_src1= (const float*)d_in[7];
  const float* a_dst1= (const float*)d_in[8];
  const float* bias1 = (const float*)d_in[9];
  const float* w2    = (const float*)d_in[10];
  const float* a_src2= (const float*)d_in[11];
  const float* a_dst2= (const float*)d_in[12];
  const float* bias2 = (const float*)d_in[13];
  const float* W_s1  = (const float*)d_in[14];
  const float* b_s1  = (const float*)d_in[15];
  const float* W_s2  = (const float*)d_in[16];
  const float* b_s2  = (const float*)d_in[17];
  const float* W_f1  = (const float*)d_in[18];
  const float* b_f1  = (const float*)d_in[19];
  const float* W_fa  = (const float*)d_in[20];
  const float* b_fa  = (const float*)d_in[21];
  const float* W_fb  = (const float*)d_in[22];
  const float* b_fb  = (const float*)d_in[23];

  // d_out layout: output1_new, output2_new, x_temp, attn_pp, attn_wp, x
  float* out     = (float*)d_out;
  float* o1n     = out;
  float* o2n     = o1n + (size_t)B_SZ * 64;
  float* xtemp   = o2n + (size_t)B_SZ * 64;
  float* attn_pp = xtemp + (size_t)B_SZ * 5;
  float* attn_wp = attn_pp + (size_t)NHEAD * NP * NP;
  float* xout    = attn_wp + (size_t)NHEAD * NALL * NALL;

  // workspace layout
  float* ws = (float*)d_ws;
  float* hp1     = ws; ws += (size_t)NHEAD * NP * HDIM;
  float* es1     = ws; ws += (size_t)NHEAD * NP;
  float* ed1     = ws; ws += (size_t)NHEAD * NP;
  float* hp2     = ws; ws += (size_t)NHEAD * NALL * HDIM;
  float* es2     = ws; ws += (size_t)NHEAD * NALL;
  float* ed2     = ws; ws += (size_t)NHEAD * NALL;
  float* h1wp    = ws; ws += (size_t)NALL * 256;
  float* h2heads = ws; ws += (size_t)NHEAD * NALL * HDIM;
  float* h2w     = ws; ws += (size_t)NW * HDIM;
  float* xraw    = ws; ws += (size_t)B_SZ * HDIM;
  float* out1    = ws; ws += (size_t)B_SZ * 128;
  float* out2    = ws; ws += (size_t)B_SZ * 128;
  float* x12     = ws; ws += (size_t)B_SZ * 256;
  float* tbuf    = ws; ws += (size_t)B_SZ * 128;

  const dim3 gblk(32 * GEMM_WAVES);   // 128 threads = 4 waves (wave32)

  // ---- GAT layer 1 ----
  // hp1[h] = pern_feature @ w1[h]    (2048x256 @ 256x32, batched over heads)
  gemm_bf16_wmma<<<dim3((NP + 63) / 64, HDIM / 32, NHEAD), gblk, 0, stream>>>(
      pern_feature, w1, nullptr, hp1, NP, 256, HDIM,
      256, HDIM, HDIM, 0L, 256L * HDIM, (long)NP * HDIM, 0);

  edge_scores_kernel<<<g1((long)NHEAD * NP, 256), 256, 0, stream>>>(
      hp1, a_src1, a_dst1, es1, ed1, NP);

  attn_softmax_kernel<<<dim3(NP), 256, 0, stream>>>(
      pern_adj, es1, ed1, attn_pp, NP);

  // h1 = elu(attn_pp @ hp1 + bias1), heads packed as columns of h1_wp rows 89..
  copy_words_kernel<<<g1((long)NW * 256, 256), 256, 0, stream>>>(word_feature, h1wp);
  gemm_bf16_wmma<<<dim3((NP + 63) / 64, HDIM / 32, NHEAD), gblk, 0, stream>>>(
      attn_pp, hp1, bias1, h1wp + (size_t)NW * 256, NP, NP, HDIM,
      NP, HDIM, 256, (long)NP * NP, (long)NP * HDIM, (long)HDIM, 2 /*elu*/);

  // ---- GAT layer 2 ----
  gemm_bf16_wmma<<<dim3((NALL + 63) / 64, HDIM / 32, NHEAD), gblk, 0, stream>>>(
      h1wp, w2, nullptr, hp2, NALL, 256, HDIM,
      256, HDIM, HDIM, 0L, 256L * HDIM, (long)NALL * HDIM, 0);

  edge_scores_kernel<<<g1((long)NHEAD * NALL, 256), 256, 0, stream>>>(
      hp2, a_src2, a_dst2, es2, ed2, NALL);

  attn_softmax_kernel<<<dim3(NALL), 256, 0, stream>>>(
      word_pern_adj, es2, ed2, attn_wp, NALL);

  gemm_bf16_wmma<<<dim3((NALL + 63) / 64, HDIM / 32, NHEAD), gblk, 0, stream>>>(
      attn_wp, hp2, bias2, h2heads, NALL, NALL, HDIM,
      NALL, HDIM, HDIM, (long)NALL * NALL, (long)NALL * HDIM, (long)NALL * HDIM, 0);

  mean_words_kernel<<<g1((long)NW * HDIM, 256), 256, 0, stream>>>(h2heads, h2w);

  // ---- user/word mixing ----
  xraw_kernel<<<g1((long)B_SZ * HDIM, 256), 256, 0, stream>>>(user_word_adj, h2w, xraw);
  xnorm_kernel<<<dim3(B_SZ), 256, 0, stream>>>(xraw, user_word_adj, xout);

  // ---- dense stack ----
  // output1 = x @ W_s1 + b_s1
  gemm_bf16_wmma<<<dim3(B_SZ / 64, 128 / 32, 1), gblk, 0, stream>>>(
      xout, W_s1, b_s1, out1, B_SZ, 256, 128, 256, 128, 128, 0L, 0L, 0L, 0);
  // output2 = sentence_embed @ W_s2 + b_s2
  gemm_bf16_wmma<<<dim3(B_SZ / 64, 128 / 32, 1), gblk, 0, stream>>>(
      sentence_emb, W_s2, b_s2, out2, B_SZ, E2D, 128, E2D, 128, 128, 0L, 0L, 0L, 0);
  // output1_new / output2_new = outputX @ W_f1 + b_f1
  gemm_bf16_wmma<<<dim3(B_SZ / 64, 64 / 32, 1), gblk, 0, stream>>>(
      out1, W_f1, b_f1, o1n, B_SZ, 128, 64, 128, 64, 64, 0L, 0L, 0L, 0);
  gemm_bf16_wmma<<<dim3(B_SZ / 64, 64 / 32, 1), gblk, 0, stream>>>(
      out2, W_f1, b_f1, o2n, B_SZ, 128, 64, 128, 64, 64, 0L, 0L, 0L, 0);
  // x_temp = relu([output1|output2] @ W_fa + b_fa) @ W_fb + b_fb
  concat_kernel<<<g1((long)B_SZ * 256, 256), 256, 0, stream>>>(out1, out2, x12);
  gemm_bf16_wmma<<<dim3(B_SZ / 64, 128 / 32, 1), gblk, 0, stream>>>(
      x12, W_fa, b_fa, tbuf, B_SZ, 256, 128, 256, 128, 128, 0L, 0L, 0L, 1 /*relu*/);
  final5_kernel<<<g1((long)B_SZ * 5, 256), 256, 0, stream>>>(tbuf, W_fb, b_fb, xtemp);
}
